// MMU_54262616818003
// MI455X (gfx1250) — matrix operand order repaired and checked by reading, not compiled, not run on hardware
//
#include <hip/hip_runtime.h>

// MMU RNN (5 steps) on gfx1250: all GEMMs via v_wmma_f32_16x16x32_bf16,
// global->LDS staging via CDNA5 async-LDS loads (ASYNCcnt), double-buffered.
// Dims: IN=512, HID=MEM=1024, OUT=512, BATCH=4096.

typedef __bf16 bf16;
typedef __attribute__((ext_vector_type(16))) __bf16 v16bf;
typedef __attribute__((ext_vector_type(8)))  float  v8f;

#define BATCH 4096

union Frag { v16bf v; uint4 q[2]; };

__device__ __forceinline__ float sigmoidf_(float x) { return 1.0f / (1.0f + __expf(-x)); }

// CDNA5 async global->LDS copy (per-lane b128), tracked by ASYNCcnt.
__device__ __forceinline__ void async_b128(unsigned lds_off, const void* gaddr) {
    asm volatile("global_load_async_to_lds_b128 %0, %1, off"
                 :: "v"(lds_off), "v"((unsigned long long)(uintptr_t)gaddr)
                 : "memory");
}
__device__ __forceinline__ void wait_async0() {
#if __has_builtin(__builtin_amdgcn_s_wait_asynccnt)
    __builtin_amdgcn_s_wait_asynccnt(0);
#else
    asm volatile("s_wait_asynccnt 0x0" ::: "memory");
#endif
}

// ---------------------------------------------------------------------------
// Generic 128x128-tile bf16 GEMM (C = A[M,Ka] * Bw[N,Ka]^T), fused epilogues.
// mode 0: gates    (bias add, sigmoid; split 4x1024 col-groups; rgmem = sig*mem)
// mode 1: decoder  (hidden = acc + b + bi*ig -> bf16)
// mode 2: encoder  (e = tanh(acc+b); mem = (1-wg)*mem + wg*e; write mem f32 + Apack bf16)
// mode 3: hid_out  (out f32 -> d_out; out bf16 -> Apack cols 512..1023)
// ---------------------------------------------------------------------------
__global__ __launch_bounds__(256) void gemm_bf16_k(
    const bf16* __restrict__ A, const bf16* __restrict__ Bw,
    const float* __restrict__ bias, int Ka, int Nout, int mode,
    void* x0, void* x1, void* x2, void* x3, void* x4)
{
    // double-buffered tiles; stride 40 bf16 = 80B (16B-aligned rows, bank-spread)
    __shared__ bf16 As[2][128][40];
    __shared__ bf16 Bs[2][128][40];

    const int tid  = threadIdx.x;
    const int lane = tid & 31;
    const int w    = tid >> 5;          // wave id 0..7
    const int wrow = w >> 1;            // 0..3 -> 32-row slabs
    const int wcol = w & 1;             // 0..1 -> 64-col slabs
    const int r    = lane & 15;
    const int h    = lane >> 4;
    const int mBase = blockIdx.y * 128;
    const int nBase = blockIdx.x * 128;

    // per-thread copy coordinates: 2 chunks of 8 bf16 per 128x32 tile
    const int row0 = tid >> 2;          // 0..63 (second chunk: +64)
    const int c8   = (tid & 3) * 8;     // 0,8,16,24

    v8f acc[2][4] = {};

    auto issue_tile = [&](int kb, int buf) {
        const bf16* ap = &A [(size_t)(mBase + row0) * Ka + kb + c8];
        const bf16* bp = &Bw[(size_t)(nBase + row0) * Ka + kb + c8];
        async_b128((unsigned)(uintptr_t)&As[buf][row0][c8],      ap);
        async_b128((unsigned)(uintptr_t)&As[buf][row0 + 64][c8], ap + (size_t)64 * Ka);
        async_b128((unsigned)(uintptr_t)&Bs[buf][row0][c8],      bp);
        async_b128((unsigned)(uintptr_t)&Bs[buf][row0 + 64][c8], bp + (size_t)64 * Ka);
    };

    const int nk = Ka >> 5;
    issue_tile(0, 0);
    wait_async0();
    __syncthreads();

    for (int kbi = 0; kbi < nk; ++kbi) {
        const int  cur  = kbi & 1;
        const bool more = (kbi + 1 < nk);
        if (more) issue_tile((kbi + 1) << 5, cur ^ 1);   // prefetch next tile async

        // A fragments: 16x32, lane half h selects K-interleave (ISA 7.12.2):
        //   elems 0..7 -> K = 8h..8h+7 ; elems 8..15 -> K = 16+8h..23+8h
        Frag a[2], b[4];
#pragma unroll
        for (int mt = 0; mt < 2; ++mt) {
            int row = wrow * 32 + mt * 16 + r;
            a[mt].q[0] = *(const uint4*)&As[cur][row][8 * h];
            a[mt].q[1] = *(const uint4*)&As[cur][row][16 + 8 * h];
        }
        // B fragments: 32x16, lane = N column; same K-interleave as A:
        //   elems 0..7 -> K = 8h..8h+7 ; elems 8..15 -> K = 16+8h..23+8h
#pragma unroll
        for (int nt = 0; nt < 4; ++nt) {
            int col = wcol * 64 + nt * 16 + r;
            b[nt].q[0] = *(const uint4*)&Bs[cur][col][8 * h];
            b[nt].q[1] = *(const uint4*)&Bs[cur][col][16 + 8 * h];
        }
#pragma unroll
        for (int mt = 0; mt < 2; ++mt)
#pragma unroll
            for (int nt = 0; nt < 4; ++nt)
                acc[mt][nt] = __builtin_amdgcn_wmma_f32_16x16x32_bf16(
                    false, a[mt].v, false, b[nt].v, (short)0, acc[mt][nt], false, false);

        if (more) {            // own async loads done -> barrier makes tile visible
            wait_async0();
            __syncthreads();
        }
    }

    // Epilogue. C/D layout: vgpr j, lanes 0..15 -> M=j, lanes 16..31 -> M=j+8; N=r.
#pragma unroll
    for (int mt = 0; mt < 2; ++mt)
#pragma unroll
        for (int nt = 0; nt < 4; ++nt) {
            int n = nBase + wcol * 64 + nt * 16 + r;
            float bv = bias ? bias[n] : 0.0f;
#pragma unroll
            for (int j = 0; j < 8; ++j) {
                int m = mBase + wrow * 32 + mt * 16 + 8 * h + j;
                float v = acc[mt][nt][j] + bv;
                if (mode == 0) {
                    float s = sigmoidf_(v);
                    int g = n >> 10, n0 = n & 1023;
                    size_t idx = (size_t)m * 1024 + n0;
                    if (g == 0)      ((bf16*)x0)[idx] = (bf16)s;                                   // block_inp
                    else if (g == 1) ((bf16*)x1)[idx] = (bf16)s;                                   // inp_gate
                    else if (g == 2) ((bf16*)x2)[idx] = (bf16)(s * ((const float*)x4)[idx]);       // read_gate*mem
                    else             ((bf16*)x3)[idx] = (bf16)s;                                   // write_gate
                } else if (mode == 1) {
                    size_t idx = (size_t)m * 1024 + n;
                    float hid = v + (float)((const bf16*)x0)[idx] * (float)((const bf16*)x1)[idx];
                    ((bf16*)x2)[idx] = (bf16)hid;                                                  // hidden
                } else if (mode == 2) {
                    size_t idx = (size_t)m * 1024 + n;
                    float e  = tanhf(v);
                    float wg = (float)((const bf16*)x0)[idx];
                    float mm = (1.0f - wg) * ((float*)x1)[idx] + wg * e;
                    ((float*)x1)[idx] = mm;                                                        // mem f32
                    ((bf16*)x2)[(size_t)m * 2048 + 1024 + n] = (bf16)mm;                           // Apack mem cols
                } else {
                    ((float*)x0)[(size_t)m * 512 + n] = v;                                         // d_out
                    ((bf16*)x1)[(size_t)m * 2048 + 512 + n] = (bf16)v;                             // Apack out cols
                }
            }
        }
}

// ---------------------------------------------------------------------------
// Prep kernels
// ---------------------------------------------------------------------------
// Packed gate weights Wg[4096][2048]: row n -> gate g=n>>10, unit j=n&1023.
// cols 0..511 = W_x[j][k], 512..1023 = W_rec[j][k-512], 1024..2047 = W_mem or 0.
__global__ void pack_gate_w(bf16* __restrict__ Wg,
    const float* Wx0, const float* Wx1, const float* Wx2, const float* Wx3,
    const float* Wr0, const float* Wr1, const float* Wr2, const float* Wr3,
    const float* Wm1, const float* Wm2, const float* Wm3)
{
    size_t idx = (size_t)blockIdx.x * blockDim.x + threadIdx.x;
    if (idx >= (size_t)4096 * 2048) return;
    int n = (int)(idx >> 11), k = (int)(idx & 2047);
    int g = n >> 10, j = n & 1023;
    float v;
    if (k < 512) {
        const float* W = g == 0 ? Wx0 : g == 1 ? Wx1 : g == 2 ? Wx2 : Wx3;
        v = W[j * 512 + k];
    } else if (k < 1024) {
        const float* W = g == 0 ? Wr0 : g == 1 ? Wr1 : g == 2 ? Wr2 : Wr3;
        v = W[j * 512 + (k - 512)];
    } else {
        if (g == 0) v = 0.0f;
        else {
            const float* W = g == 1 ? Wm1 : g == 2 ? Wm2 : Wm3;
            v = W[j * 1024 + (k - 1024)];
        }
    }
    Wg[idx] = (bf16)v;
}

__global__ void pack_bias_k(float* bg, float* bdec, float* benc,
    const float* b_inp, const float* b_rec_inp,
    const float* b_inpgate, const float* b_rec_inpgate, const float* b_mem_inpgate,
    const float* b_readgate, const float* b_rec_readgate, const float* b_mem_readgate,
    const float* b_writegate, const float* b_rec_writegate, const float* b_mem_writegate,
    const float* b_decoder, const float* b_encoder)
{
    int i = blockIdx.x * blockDim.x + threadIdx.x;
    if (i < 4096) {
        int g = i >> 10, j = i & 1023;
        float v;
        if (g == 0)      v = b_inp[j] + b_rec_inp[j];
        else if (g == 1) v = b_inpgate[j] + b_rec_inpgate[j] + b_mem_inpgate[j];
        else if (g == 2) v = b_readgate[j] + b_rec_readgate[j] + b_mem_readgate[j];
        else             v = b_writegate[j] + b_rec_writegate[j] + b_mem_writegate[j];
        bg[i] = v;
    } else if (i < 5120) {
        bdec[i - 4096] = b_decoder[i - 4096];
    } else if (i < 6144) {
        benc[i - 5120] = b_encoder[i - 5120];
    }
}

__global__ void f32_to_bf16_k(bf16* dst, const float* src, int n) {
    int i = blockIdx.x * blockDim.x + threadIdx.x;
    if (i < n) dst[i] = (bf16)src[i];
}

// Step 4 only: x = input.T ; input is [512, 4096] row-major.
__global__ void fill_x_k(bf16* __restrict__ Apack, const float* __restrict__ input) {
    int idx = blockIdx.x * blockDim.x + threadIdx.x;
    if (idx >= 4096 * 512) return;
    int b = idx >> 9, k = idx & 511;
    Apack[(size_t)b * 2048 + k] = (bf16)input[(size_t)k * 4096 + b];
}

// ---------------------------------------------------------------------------
extern "C" void kernel_launch(void* const* d_in, const int* in_sizes, int n_in,
                              void* d_out, int out_size, void* d_ws, size_t ws_size,
                              hipStream_t stream)
{
    (void)in_sizes; (void)n_in; (void)out_size; (void)ws_size;
    const float* input          = (const float*)d_in[0];
    const float* W_inpgate      = (const float*)d_in[1];
    const float* b_inpgate      = (const float*)d_in[2];
    const float* W_rec_inpgate  = (const float*)d_in[3];
    const float* b_rec_inpgate  = (const float*)d_in[4];
    const float* W_mem_inpgate  = (const float*)d_in[5];
    const float* b_mem_inpgate  = (const float*)d_in[6];
    const float* W_inp          = (const float*)d_in[7];
    const float* b_inp          = (const float*)d_in[8];
    const float* W_rec_inp      = (const float*)d_in[9];
    const float* b_rec_inp      = (const float*)d_in[10];
    const float* W_readgate     = (const float*)d_in[11];
    const float* b_readgate     = (const float*)d_in[12];
    const float* W_rec_readgate = (const float*)d_in[13];
    const float* b_rec_readgate = (const float*)d_in[14];
    const float* W_mem_readgate = (const float*)d_in[15];
    const float* b_mem_readgate = (const float*)d_in[16];
    const float* W_decoder      = (const float*)d_in[17];
    const float* b_decoder      = (const float*)d_in[18];
    const float* W_writegate    = (const float*)d_in[19];
    const float* b_writegate    = (const float*)d_in[20];
    const float* W_rec_writegate= (const float*)d_in[21];
    const float* b_rec_writegate= (const float*)d_in[22];
    const float* W_mem_writegate= (const float*)d_in[23];
    const float* b_mem_writegate= (const float*)d_in[24];
    const float* W_encoder      = (const float*)d_in[25];
    const float* b_encoder      = (const float*)d_in[26];
    const float* w_hid_out      = (const float*)d_in[27];

    // ---- workspace carve (256B aligned) ----
    char* ws = (char*)d_ws;
    size_t off = 0;
    auto carve = [&](size_t bytes) { void* p = ws + off; off += (bytes + 255) & ~(size_t)255; return p; };
    bf16*  Wg     = (bf16*) carve((size_t)4096 * 2048 * 2);   // packed gate weights
    float* bg     = (float*)carve((size_t)4096 * 4);          // packed gate bias
    bf16*  Wdec   = (bf16*) carve((size_t)1024 * 1024 * 2);
    bf16*  Wenc   = (bf16*) carve((size_t)1024 * 1024 * 2);
    bf16*  Whid   = (bf16*) carve((size_t)512 * 1024 * 2);
    float* bdec   = (float*)carve((size_t)1024 * 4);
    float* benc   = (float*)carve((size_t)1024 * 4);
    bf16*  Apack  = (bf16*) carve((size_t)BATCH * 2048 * 2);  // [x | out | mem] bf16
    bf16*  bi     = (bf16*) carve((size_t)BATCH * 1024 * 2);  // sigmoid(block_inp)
    bf16*  ig     = (bf16*) carve((size_t)BATCH * 1024 * 2);  // inp_gate
    bf16*  wgb    = (bf16*) carve((size_t)BATCH * 1024 * 2);  // write_gate
    bf16*  rgmem  = (bf16*) carve((size_t)BATCH * 1024 * 2);  // read_gate * mem
    bf16*  hidden = (bf16*) carve((size_t)BATCH * 1024 * 2);
    float* memf   = (float*)carve((size_t)BATCH * 1024 * 4);  // mem state f32

    // ---- init: zero Apack (x=0 for steps 0..3; out/mem cols 0) and mem ----
    hipMemsetAsync(Apack, 0, (size_t)BATCH * 2048 * 2, stream);
    hipMemsetAsync(memf,  0, (size_t)BATCH * 1024 * 4, stream);

    // ---- prep: pack weights/biases to bf16 ----
    {
        size_t tot = (size_t)4096 * 2048;
        pack_gate_w<<<dim3((unsigned)((tot + 255) / 256)), dim3(256), 0, stream>>>(
            Wg,
            W_inp, W_inpgate, W_readgate, W_writegate,
            W_rec_inp, W_rec_inpgate, W_rec_readgate, W_rec_writegate,
            W_mem_inpgate, W_mem_readgate, W_mem_writegate);
        pack_bias_k<<<dim3(24), dim3(256), 0, stream>>>(
            bg, bdec, benc,
            b_inp, b_rec_inp,
            b_inpgate, b_rec_inpgate, b_mem_inpgate,
            b_readgate, b_rec_readgate, b_mem_readgate,
            b_writegate, b_rec_writegate, b_mem_writegate,
            b_decoder, b_encoder);
        int n1 = 1024 * 1024;
        f32_to_bf16_k<<<dim3((n1 + 255) / 256), dim3(256), 0, stream>>>(Wdec, W_decoder, n1);
        f32_to_bf16_k<<<dim3((n1 + 255) / 256), dim3(256), 0, stream>>>(Wenc, W_encoder, n1);
        int n2 = 512 * 1024;
        f32_to_bf16_k<<<dim3((n2 + 255) / 256), dim3(256), 0, stream>>>(Whid, w_hid_out, n2);
    }

    auto gemm = [&](const bf16* A, const bf16* Bw, const float* bias, int Ka, int Nout,
                    int mode, void* x0, void* x1, void* x2, void* x3, void* x4) {
        dim3 grid(Nout / 128, BATCH / 128);
        gemm_bf16_k<<<grid, dim3(256), 0, stream>>>(A, Bw, bias, Ka, Nout, mode, x0, x1, x2, x3, x4);
    };

    // ---- 5 recurrent steps (x = 0 for steps 0..3, x = input.T at step 4) ----
    for (int s = 0; s < 5; ++s) {
        if (s == 4)
            fill_x_k<<<dim3((4096 * 512 + 255) / 256), dim3(256), 0, stream>>>(Apack, input);
        // all 4 gates in one fused GEMM: [4096 x 2048] x [4096 x 2048]^T
        gemm(Apack, Wg, bg, 2048, 4096, 0, bi, ig, rgmem, wgb, memf);
        // hidden = decoder(read_gate*mem) + block_inp*inp_gate
        gemm(rgmem, Wdec, bdec, 1024, 1024, 1, bi, ig, hidden, nullptr, nullptr);
        // encoded = tanh(encoder(hidden)); mem update; Apack mem cols refresh
        gemm(hidden, Wenc, benc, 1024, 1024, 2, wgb, memf, Apack, nullptr, nullptr);
        // out = hidden @ w_hid_out^T -> d_out (f32) + Apack out cols (bf16)
        gemm(hidden, Whid, nullptr, 1024, 512, 3, d_out, Apack, nullptr, nullptr, nullptr);
    }
}
